// BasicModel_43387759624704
// MI455X (gfx1250) — matrix operand order repaired and checked by reading, not compiled, not run on hardware
//
#include <hip/hip_runtime.h>
#include <hip/hip_bf16.h>
#include <math.h>

// ---------------------------------------------------------------------------
// Problem constants (from reference): T=512, B=256, COV=128, H=512, L=2
// ---------------------------------------------------------------------------
#define TT_  512
#define BB_  256
#define COV_ 128
#define HH_  512
#define G3H_ 1536          // 3*H
#define K0P_ 160           // layer0 x-part K (129) padded to mult of 32
#define KW0_ 672           // packed W rows layer0: 160 + 512
#define KW1_ 1024          // packed W rows layer1: 512 + 512
#define NSUB_ (G3H_ / 16)  // 96 n-subtiles across the 3 gates
#define MT_   (BB_ / 16)   // 16 batch m-tiles
#define HKC_  (HH_ / 32)   // 16 k-chunks over H
#define XKC_  (K0P_ / 32)  // 5 k-chunks over padded x
#define HROWS_ (BB_ * (TT_ - 1))   // 130816 head rows
#define HMT_   (HROWS_ / 16)       // 8176 head m-tiles

typedef __attribute__((ext_vector_type(16))) __bf16 v16bf;
typedef __attribute__((ext_vector_type(8)))  float  v8f;

// Fragment layouts (per the CDNA5 ISA VGPR tables, wave32):
//  A frag (16x32, M=lane%16):  elem e -> K = (e<8 ? e+8*half : e+8+8*half)
//  B frag (32x16, N=lane%16):  elem e -> K = (e<8 ? e+8*half : e+8+8*half)
// Buffers are stored as [kchunk][tile][lane(32)][elem(16)] bf16 so one lane's
// fragment is 32 contiguous bytes (2x global_load_b128, coalesced per wave).

// inverse of the A-frag map: (M,K) -> flat index in [kchunk][mtile][32][16]
__device__ __forceinline__ long afrag_idx(int mtiles, int M, int K) {
    int mt = M >> 4, Ml = M & 15;
    int kc = K >> 5, kin = K & 31;
    int hf = (kin >> 3) & 1;
    int e  = (kin & 7) + ((kin >> 4) << 3);
    int ln = (hf << 4) + Ml;
    return ((long)(kc * mtiles + mt) * 32 + ln) * 16 + e;
}

// ---------------------------------------------------------------------------
// Packing kernels (run once, cheap)
// ---------------------------------------------------------------------------
// xbp[t][kc(5)][mt(16)][lane][e] in A-frag order
__global__ void pack_x_frag_kernel(const float* __restrict__ x, __bf16* __restrict__ xbp) {
    long i = (long)blockIdx.x * blockDim.x + threadIdx.x;   // over T * 40960
    if (i >= (long)TT_ * XKC_ * MT_ * 32 * 16) return;
    int  w  = (int)(i % (XKC_ * MT_ * 32 * 16));
    int  t  = (int)(i / (XKC_ * MT_ * 32 * 16));
    int  e  = w & 15;
    int  L  = (w >> 4) & 31;
    int  mt = (w >> 9) & 15;
    int  kc = w >> 13;
    int  hf = L >> 4;
    int  kin = (e < 8) ? (e + 8 * hf) : (e + 8 + 8 * hf);
    int  k  = kc * 32 + kin;
    int  b  = mt * 16 + (L & 15);
    float v = (k < COV_ + 1) ? x[((long)t * BB_ + b) * (COV_ + 1) + k] : 0.0f;
    xbp[i] = (__bf16)v;
}

// Bp0[kc(21)][ns(96)][lane][e] in B-frag order; rows: [Wih0 | pad | Whh0]
__global__ void pack_w0_frag_kernel(const float* __restrict__ Wih, const float* __restrict__ Whh,
                                    __bf16* __restrict__ Bp) {
    long i = (long)blockIdx.x * blockDim.x + threadIdx.x;   // over KW0_*1536
    if (i >= (long)KW0_ * G3H_) return;
    int e  = (int)(i & 15);
    int L  = (int)((i >> 4) & 31);
    int ns = (int)((i >> 9) % NSUB_);
    int kc = (int)(i / ((long)NSUB_ * 32 * 16));
    int k  = kc * 32 + ((e < 8) ? (e + 8 * (L >> 4)) : (e + 8 + 8 * (L >> 4)));
    int n  = ns * 16 + (L & 15);
    float v;
    if (k < COV_ + 1)  v = Wih[(long)n * (COV_ + 1) + k];
    else if (k < K0P_) v = 0.0f;
    else               v = Whh[(long)n * HH_ + (k - K0P_)];
    Bp[i] = (__bf16)v;
}

// Bp1[kc(32)][ns(96)][lane][e]; rows: [Wih1 | Whh1]
__global__ void pack_w1_frag_kernel(const float* __restrict__ Wih, const float* __restrict__ Whh,
                                    __bf16* __restrict__ Bp) {
    long i = (long)blockIdx.x * blockDim.x + threadIdx.x;   // over KW1_*1536
    if (i >= (long)KW1_ * G3H_) return;
    int e  = (int)(i & 15);
    int L  = (int)((i >> 4) & 31);
    int ns = (int)((i >> 9) % NSUB_);
    int kc = (int)(i / ((long)NSUB_ * 32 * 16));
    int k  = kc * 32 + ((e < 8) ? (e + 8 * (L >> 4)) : (e + 8 + 8 * (L >> 4)));
    int n  = ns * 16 + (L & 15);
    float v = (k < HH_) ? Wih[(long)n * HH_ + k]
                        : Whh[(long)n * HH_ + (k - HH_)];
    Bp[i] = (__bf16)v;
}

// Bc[kc(16)][ns(8)][lane][e] from W_cov[c][k], k<512
__global__ void pack_wc_frag_kernel(const float* __restrict__ Wcov, __bf16* __restrict__ Bc) {
    int i = blockIdx.x * blockDim.x + threadIdx.x;          // over 512*128
    if (i >= HH_ * COV_) return;
    int e  = i & 15;
    int L  = (i >> 4) & 31;
    int ns = (i >> 9) & 7;
    int kc = i >> 12;
    int k  = kc * 32 + ((e < 8) ? (e + 8 * (L >> 4)) : (e + 8 + 8 * (L >> 4)));
    int c  = ns * 16 + (L & 15);
    Bc[i] = (__bf16)Wcov[(long)c * (HH_ + 1) + k];
}

// h0 -> row-major f32 carry + A-frag bf16 copies for both layers
__global__ void init_h_kernel(const float* __restrict__ h0,
                              float* __restrict__ hf0, __bf16* __restrict__ hbp0,
                              float* __restrict__ hf1, __bf16* __restrict__ hbp1) {
    int i = blockIdx.x * blockDim.x + threadIdx.x;          // over 2*B*H (frag coords)
    if (i >= 2 * BB_ * HH_) return;
    int layer = i >> 17;
    int w  = i & (BB_ * HH_ - 1);
    int e  = w & 15;
    int L  = (w >> 4) & 31;
    int mt = (w >> 9) & 15;
    int kc = w >> 13;
    int hf = L >> 4;
    int kin = (e < 8) ? (e + 8 * hf) : (e + 8 + 8 * hf);
    int k  = kc * 32 + kin;
    int b  = mt * 16 + (L & 15);
    float v = h0[(long)layer * BB_ * HH_ + (long)b * HH_ + k];
    if (layer == 0) { hf0[(long)b * HH_ + k] = v; hbp0[w] = (__bf16)v; }
    else            { hf1[(long)b * HH_ + k] = v; hbp1[w] = (__bf16)v; }
}

// ---------------------------------------------------------------------------
// Per-timestep fused GRU kernel (LDS-free, all operands pre-packed as frags):
//   gates = [A0 | h_prev] @ Wpack  via v_wmma_f32_16x16x32_bf16, f32 accum,
//   then sigmoid/tanh GRU update + ragged-length masking, fused.
// Workgroup = 128 threads (4 wave32s), tile = 32 batch x 32 hidden cols.
// omode: 0 -> write o in layer-A-frag layout (feeds next layer)
//        1 -> write o in head-frag layout at rows R = b*511 + t  (t<511)
// ---------------------------------------------------------------------------
__global__ __launch_bounds__(128)
void gru_step_kernel(const __bf16* __restrict__ A0p, int kc0,
                     const __bf16* __restrict__ hbp,
                     const __bf16* __restrict__ Bp,
                     const float* __restrict__ bih, const float* __restrict__ bhh,
                     const float* __restrict__ hin,
                     const int* __restrict__ lengths, int t,
                     float* __restrict__ hout, __bf16* __restrict__ hbout,
                     __bf16* __restrict__ oout, int omode) {
    const int tid  = threadIdx.x;
    const int lane = tid & 31;
    const int wave = tid >> 5;
    const int half = lane >> 4;
    const int m0w  = (wave >> 1) << 4;                  // 0 / 16
    const int n0w  = (wave & 1) << 4;                   // 0 / 16
    const int mtile = blockIdx.x * 2 + (wave >> 1);     // of 16
    const int nsub  = blockIdx.y * 2 + (wave & 1);      // of 32 (within one gate)

    const v16bf* Ap0 = (const v16bf*)A0p;
    const v16bf* Ap1 = (const v16bf*)hbp;
    const v16bf* Bpv = (const v16bf*)Bp;

    v8f racc = {}; v8f zacc = {}; v8f iacc = {}; v8f hacc = {};

    // segment 0: x-part (or o0-part). n-gate accumulates separately (iacc).
    for (int kc = 0; kc < kc0; ++kc) {
        v16bf a  = Ap0[(kc * MT_ + mtile) * 32 + lane];
        v16bf b0 = Bpv[((long)(kc * NSUB_) + 0 * 32 + nsub) * 32 + lane];
        v16bf b1 = Bpv[((long)(kc * NSUB_) + 1 * 32 + nsub) * 32 + lane];
        v16bf b2 = Bpv[((long)(kc * NSUB_) + 2 * 32 + nsub) * 32 + lane];
        racc = __builtin_amdgcn_wmma_f32_16x16x32_bf16(false, a, false, b0, (short)0, racc, false, false);
        zacc = __builtin_amdgcn_wmma_f32_16x16x32_bf16(false, a, false, b1, (short)0, zacc, false, false);
        iacc = __builtin_amdgcn_wmma_f32_16x16x32_bf16(false, a, false, b2, (short)0, iacc, false, false);
    }
    // segment 1: h-part. n-gate h-contribution accumulates into hacc.
    for (int kc = 0; kc < HKC_; ++kc) {
        int ka = kc0 + kc;
        v16bf a  = Ap1[(kc * MT_ + mtile) * 32 + lane];
        v16bf b0 = Bpv[((long)(ka * NSUB_) + 0 * 32 + nsub) * 32 + lane];
        v16bf b1 = Bpv[((long)(ka * NSUB_) + 1 * 32 + nsub) * 32 + lane];
        v16bf b2 = Bpv[((long)(ka * NSUB_) + 2 * 32 + nsub) * 32 + lane];
        racc = __builtin_amdgcn_wmma_f32_16x16x32_bf16(false, a, false, b0, (short)0, racc, false, false);
        zacc = __builtin_amdgcn_wmma_f32_16x16x32_bf16(false, a, false, b1, (short)0, zacc, false, false);
        hacc = __builtin_amdgcn_wmma_f32_16x16x32_bf16(false, a, false, b2, (short)0, hacc, false, false);
    }

    // fused GRU nonlinearity + ragged-length carry (C layout: row = v + 8*half)
    const int col  = blockIdx.y * 32 + n0w + (lane & 15);
    const float br  = bih[col] + bhh[col];
    const float bz  = bih[HH_ + col] + bhh[HH_ + col];
    const float bin = bih[2 * HH_ + col];
    const float bhn = bhh[2 * HH_ + col];
    #pragma unroll
    for (int v = 0; v < 8; ++v) {
        int row = blockIdx.x * 32 + m0w + 8 * half + v;   // batch index
        float hp = hin[(long)row * HH_ + col];
        float r  = 1.0f / (1.0f + __expf(-(racc[v] + br)));
        float z  = 1.0f / (1.0f + __expf(-(zacc[v] + bz)));
        float nn = tanhf(iacc[v] + bin + r * (hacc[v] + bhn));
        float hn = (1.0f - z) * nn + z * hp;
        bool valid = t < lengths[row];
        float hc = valid ? hn : hp;                        // freeze carry past length
        float ov = valid ? hn : 0.0f;                      // zero-padded output
        hout[(long)row * HH_ + col] = hc;
        hbout[afrag_idx(MT_, row, col)] = (__bf16)hc;      // next-step A frag
        if (omode == 0) {
            oout[afrag_idx(MT_, row, col)] = (__bf16)ov;   // layer-1 A frag (slab for t)
        } else if (t < TT_ - 1) {
            int R = row * (TT_ - 1) + t;                   // head row
            oout[afrag_idx(HMT_, R, col)] = (__bf16)ov;
        }
    }
}

// ---------------------------------------------------------------------------
// Head GEMM: preds[b,t,c] = o1[t,b,:] . Wcov[c,:512]  (bf16 WMMA, K=512)
//            + time * Wcov[c,512] + bcov[c]   (exact f32 scalar part)
// Rows = (b,t) pairs, M = 130816 (4088 row tiles of 32), N = 128. LDS-free.
// ---------------------------------------------------------------------------
__global__ __launch_bounds__(128)
void head_kernel(const __bf16* __restrict__ o1p, const __bf16* __restrict__ Bcp,
                 const float* __restrict__ x, const float* __restrict__ Wcov,
                 const float* __restrict__ bcov, const int* __restrict__ lengths,
                 float* __restrict__ out) {
    const int tid  = threadIdx.x;
    const int lane = tid & 31;
    const int wave = tid >> 5;
    const int half = lane >> 4;
    const int m0w  = (wave >> 1) << 4;
    const int n0w  = (wave & 1) << 4;
    const int mtile = blockIdx.x * 2 + (wave >> 1);     // of 8176
    const int nsub  = blockIdx.y * 2 + (wave & 1);      // of 8

    const v16bf* Ap = (const v16bf*)o1p;
    const v16bf* Bv = (const v16bf*)Bcp;

    v8f acc = {};
    for (int kc = 0; kc < HKC_; ++kc) {
        v16bf a = Ap[((long)kc * HMT_ + mtile) * 32 + lane];
        v16bf b = Bv[(kc * 8 + nsub) * 32 + lane];
        acc = __builtin_amdgcn_wmma_f32_16x16x32_bf16(false, a, false, b, (short)0, acc, false, false);
    }

    const int c = blockIdx.y * 32 + n0w + (lane & 15);
    const float wtime = Wcov[(long)c * (HH_ + 1) + HH_];
    const float bc = bcov[c];
    #pragma unroll
    for (int v = 0; v < 8; ++v) {
        int R = blockIdx.x * 32 + m0w + 8 * half + v;
        int b = R / (TT_ - 1);
        int tt = R - b * (TT_ - 1);
        float time = x[((long)tt * BB_ + b) * (COV_ + 1)];   // x[t,b,0]
        float p = acc[v] + time * wtime + bc;
        bool valid = tt < lengths[b] - 1;
        out[((long)b * (TT_ - 1) + tt) * (COV_ + 1) + c] = valid ? p : 0.0f;
    }
}

// last output channel stays zero in the reference
__global__ void zero_last_channel_kernel(float* __restrict__ out) {
    int i = blockIdx.x * blockDim.x + threadIdx.x;
    if (i >= HROWS_) return;
    out[(long)i * (COV_ + 1) + COV_] = 0.0f;
}

// dist = exp(-(last_flat @ W_par^T + b_par)) with torch-.view flattening of [L,B,H]
__global__ void dist_kernel(const float* __restrict__ hl0, const float* __restrict__ hl1,
                            const float* __restrict__ Wpar, const float* __restrict__ bpar,
                            float* __restrict__ dout) {
    int i = blockIdx.x * blockDim.x + threadIdx.x;   // over B*2
    if (i >= BB_ * 2) return;
    int b = i >> 1, p = i & 1;
    float s = bpar[p];
    for (int j = 0; j < 2 * HH_; ++j) {
        long f = (long)b * (2 * HH_) + j;            // flat index into [2,256,512]
        const float* hl = (f >> 17) ? hl1 : hl0;     // / (256*512)
        int bb = (int)((f >> 9) & 255);
        int h  = (int)(f & 511);
        s += Wpar[(long)p * (2 * HH_) + j] * hl[(long)bb * HH_ + h];
    }
    dout[(long)b * 2 + p] = __expf(-s);
}

// ---------------------------------------------------------------------------
// Host side
// ---------------------------------------------------------------------------
extern "C" void kernel_launch(void* const* d_in, const int* in_sizes, int n_in,
                              void* d_out, int out_size, void* d_ws, size_t ws_size,
                              hipStream_t stream) {
    const float* x       = (const float*)d_in[0];
    const int*   lengths = (const int*)d_in[1];
    const float* h0      = (const float*)d_in[2];
    const float* Wih0    = (const float*)d_in[3];
    const float* Whh0    = (const float*)d_in[4];
    const float* bih0    = (const float*)d_in[5];
    const float* bhh0    = (const float*)d_in[6];
    const float* Wih1    = (const float*)d_in[7];
    const float* Whh1    = (const float*)d_in[8];
    const float* bih1    = (const float*)d_in[9];
    const float* bhh1    = (const float*)d_in[10];
    const float* Wcov    = (const float*)d_in[11];
    const float* bcov    = (const float*)d_in[12];
    const float* Wpar    = (const float*)d_in[13];
    const float* bpar    = (const float*)d_in[14];
    float* out = (float*)d_out;

    // workspace carve-out
    char* ws = (char*)d_ws;
    size_t off = 0;
    auto carve = [&](size_t bytes) -> void* {
        off = (off + 255) & ~(size_t)255;
        void* p = ws + off;
        off += bytes;
        return p;
    };
    __bf16* xbp  = (__bf16*)carve((size_t)TT_ * BB_ * K0P_ * 2);        //  42 MB
    __bf16* Bp0  = (__bf16*)carve((size_t)KW0_ * G3H_ * 2);             //   2 MB
    __bf16* Bp1  = (__bf16*)carve((size_t)KW1_ * G3H_ * 2);             //   3 MB
    __bf16* Bcp  = (__bf16*)carve((size_t)HH_ * COV_ * 2);              // 128 KB
    __bf16* o0bp = (__bf16*)carve((size_t)TT_ * BB_ * HH_ * 2);         // 134 MB
    __bf16* o1p  = (__bf16*)carve((size_t)HKC_ * HMT_ * 32 * 16 * 2);   // 134 MB
    float*  hf0[2]; float* hf1[2]; __bf16* hb0[2]; __bf16* hb1[2];
    for (int i = 0; i < 2; ++i) {
        hf0[i] = (float*)carve((size_t)BB_ * HH_ * 4);
        hf1[i] = (float*)carve((size_t)BB_ * HH_ * 4);
        hb0[i] = (__bf16*)carve((size_t)BB_ * HH_ * 2);
        hb1[i] = (__bf16*)carve((size_t)BB_ * HH_ * 2);
    }

    // ---- fragment packing ----
    {
        long n = (long)TT_ * BB_ * K0P_;
        pack_x_frag_kernel<<<(unsigned)((n + 255) / 256), 256, 0, stream>>>(x, xbp);
    }
    {
        long n = (long)KW0_ * G3H_;
        pack_w0_frag_kernel<<<(unsigned)((n + 255) / 256), 256, 0, stream>>>(Wih0, Whh0, Bp0);
    }
    {
        long n = (long)KW1_ * G3H_;
        pack_w1_frag_kernel<<<(unsigned)((n + 255) / 256), 256, 0, stream>>>(Wih1, Whh1, Bp1);
    }
    pack_wc_frag_kernel<<<(HH_ * COV_ + 255) / 256, 256, 0, stream>>>(Wcov, Bcp);
    init_h_kernel<<<(2 * BB_ * HH_ + 255) / 256, 256, 0, stream>>>(h0, hf0[0], hb0[0], hf1[0], hb1[0]);

    // ---- layer 0 scan: 512 fused GEMM+GRU steps (ping-pong h buffers) ----
    dim3 sgrid(BB_ / 32, HH_ / 32);   // 8 x 16 workgroups, 128 thr (4 wave32)
    const size_t xslab = (size_t)XKC_ * MT_ * 32 * 16;   // 40960 elems per t
    const size_t oslab = (size_t)HKC_ * MT_ * 32 * 16;   // 131072 elems per t
    for (int t = 0; t < TT_; ++t) {
        int c = t & 1;
        gru_step_kernel<<<sgrid, 128, 0, stream>>>(
            xbp + (size_t)t * xslab, XKC_,
            hb0[c], Bp0, bih0, bhh0, hf0[c], lengths, t,
            hf0[1 - c], hb0[1 - c], o0bp + (size_t)t * oslab, /*omode=*/0);
    }
    // ---- layer 1 scan (output written directly in head-frag layout) ----
    for (int t = 0; t < TT_; ++t) {
        int c = t & 1;
        gru_step_kernel<<<sgrid, 128, 0, stream>>>(
            o0bp + (size_t)t * oslab, HKC_,
            hb1[c], Bp1, bih1, bhh1, hf1[c], lengths, t,
            hf1[1 - c], hb1[1 - c], o1p, /*omode=*/1);
    }

    // ---- prediction head ----
    head_kernel<<<dim3(HROWS_ / 32, COV_ / 32), 128, 0, stream>>>(
        o1p, Bcp, x, Wcov, bcov, lengths, out);
    zero_last_channel_kernel<<<(HROWS_ + 255) / 256, 256, 0, stream>>>(out);

    // ---- distribution params (final h after 512 steps lives in buffer 0) ----
    dist_kernel<<<2, 256, 0, stream>>>(hf0[0], hf1[0], Wpar, bpar,
                                       out + (size_t)HROWS_ * (COV_ + 1));
}
